// ReferenceMoEBlock_37933151158594
// MI455X (gfx1250) — matrix operand order repaired and checked by reading, not compiled, not run on hardware
//
#include <hip/hip_runtime.h>
#include <hip/hip_bf16.h>
#include <stdint.h>

// Problem constants (reference: B=4,S=1024,D=1024, I=512, E=32, TOP_K=8, 8 groups, top-4 groups)
#define TT      4096
#define DD      1024
#define II      512
#define EE      32
#define TOPK    8
#define NGROUP  8
#define TOPKG   4
#define MBLK    32          // tokens per FFN workgroup

typedef __attribute__((ext_vector_type(16))) __bf16 v16bf;
typedef __attribute__((ext_vector_type(8)))  __bf16 v8bf;
typedef __attribute__((ext_vector_type(8)))  float  v8f;
typedef __attribute__((ext_vector_type(2)))  float  v2f;

__device__ __forceinline__ __bf16 f2bf(float f) { return (__bf16)f; }  // v_cvt, RNE

// ---------------------------------------------------------------------------
// Bulk fp32 -> bf16 weight conversion (one-time per launch, HBM-bound)
// ---------------------------------------------------------------------------
__global__ __launch_bounds__(256) void cvt_bf16_kernel(
    const float* __restrict__ src, __bf16* __restrict__ dst, int n4)
{
  const int i = blockIdx.x * 256 + threadIdx.x;
  if (i >= n4) return;
  const float4 v = ((const float4*)src)[i];
  __bf16* d = dst + (size_t)i * 4;
  d[0] = f2bf(v.x); d[1] = f2bf(v.y); d[2] = f2bf(v.z); d[3] = f2bf(v.w);
}

// ---------------------------------------------------------------------------
// Router: logits = X @ gate_w^T (fp32 WMMA), sigmoid, group top-k, combine wts
// One wave (32 threads) per 16-token tile; two 16-expert column tiles.
// ---------------------------------------------------------------------------
__global__ __launch_bounds__(32) void router_kernel(
    const float* __restrict__ x,        // [T, D]
    const float* __restrict__ gw,       // [E, D]
    float* __restrict__ logits,         // [T, E]
    float* __restrict__ combine)        // [T, E]
{
  const int tile = blockIdx.x;
  const int lane = threadIdx.x;
  const int nl = lane & 15;
  const int hl = lane >> 4;
  __shared__ float sc[16][EE];

#if __has_builtin(__builtin_amdgcn_wmma_f32_16x16x4_f32)
  // A 16x4 f32: lane l holds row l%16, elements K = e + 2*(l/16)
  // B 4x16  f32: lane l holds col l%16, elements K = e + 2*(l/16)
  v8f acc0 = {}; v8f acc1 = {};
  const float* xr = x  + (size_t)(tile * 16 + nl) * DD;
  const float* g0 = gw + (size_t)nl * DD;
  const float* g1 = gw + (size_t)(16 + nl) * DD;
  for (int k0 = 0; k0 < DD; k0 += 4) {
    const int k = k0 + 2 * hl;
    v2f a;  a.x  = xr[k]; a.y  = xr[k + 1];
    v2f b0; b0.x = g0[k]; b0.y = g0[k + 1];
    v2f b1; b1.x = g1[k]; b1.y = g1[k + 1];
    acc0 = __builtin_amdgcn_wmma_f32_16x16x4_f32(false, a, false, b0, (short)0, acc0, false, false);
    acc1 = __builtin_amdgcn_wmma_f32_16x16x4_f32(false, a, false, b1, (short)0, acc1, false, false);
  }
  // C/D layout: vgpr r -> M = r + 8*(lane/16), N = lane%16
#pragma unroll
  for (int r = 0; r < 8; ++r) {
    const int m = r + 8 * hl;
    const size_t t = (size_t)(tile * 16 + m);
    logits[t * EE + nl]      = acc0[r];
    logits[t * EE + 16 + nl] = acc1[r];
    sc[m][nl]      = 1.f / (1.f + __expf(-acc0[r]));
    sc[m][16 + nl] = 1.f / (1.f + __expf(-acc1[r]));
  }
#else
  if (lane < 16) {
    const float* xrow = x + (size_t)(tile * 16 + lane) * DD;
    for (int e = 0; e < EE; ++e) {
      const float* grow = gw + (size_t)e * DD;
      float dot = 0.f;
      for (int k = 0; k < DD; ++k) dot += xrow[k] * grow[k];
      logits[(size_t)(tile * 16 + lane) * EE + e] = dot;
      sc[lane][e] = 1.f / (1.f + __expf(-dot));
    }
  }
#endif
  __syncthreads();

  if (lane < 16) {
    const size_t t = (size_t)(tile * 16 + lane);
    float s[EE];
#pragma unroll
    for (int i = 0; i < EE; ++i) s[i] = sc[lane][i];

    float gs[NGROUP];
#pragma unroll
    for (int g = 0; g < NGROUP; ++g)
      gs[g] = s[4 * g] + s[4 * g + 1] + s[4 * g + 2] + s[4 * g + 3];

    bool gsel[NGROUP];
#pragma unroll
    for (int g = 0; g < NGROUP; ++g) gsel[g] = false;
    for (int j = 0; j < TOPKG; ++j) {          // top-4 groups, first-on-tie
      int best = 0; float bv = -1e30f;
      for (int g = 0; g < NGROUP; ++g)
        if (!gsel[g] && gs[g] > bv) { bv = gs[g]; best = g; }
      gsel[best] = true;
    }

    float ms[EE];
#pragma unroll
    for (int i = 0; i < EE; ++i) ms[i] = gsel[i >> 2] ? s[i] : 0.f;

    float comb[EE];
#pragma unroll
    for (int i = 0; i < EE; ++i) comb[i] = 0.f;
    int   idxs[TOPK]; float wv[TOPK]; float wsum = 0.f;
    for (int j = 0; j < TOPK; ++j) {           // top-8 experts, first-on-tie
      int best = 0; float bv = -1.f;
      for (int i = 0; i < EE; ++i)
        if (ms[i] > bv) { bv = ms[i]; best = i; }
      idxs[j] = best; wv[j] = bv; wsum += bv; ms[best] = -2.f;
    }
    const float inv = 1.f / fmaxf(wsum, 1e-12f);
    for (int j = 0; j < TOPK; ++j) comb[idxs[j]] += wv[j] * inv;
#pragma unroll
    for (int i = 0; i < EE; ++i) combine[t * EE + i] = comb[i];
  }
}

// ---------------------------------------------------------------------------
// Build per-expert token lists (gather-based sparse MoE)
// ---------------------------------------------------------------------------
__global__ __launch_bounds__(256) void build_lists_kernel(
    const float* __restrict__ combine, int* __restrict__ counts, int* __restrict__ lists)
{
  const int t = blockIdx.x * blockDim.x + threadIdx.x;
  if (t >= TT) return;
  for (int e = 0; e < EE; ++e) {
    if (combine[(size_t)t * EE + e] != 0.f) {
      const int pos = atomicAdd(&counts[e], 1);
      lists[e * TT + pos] = t;
    }
  }
}

// ---------------------------------------------------------------------------
// Fragment helpers.
// A 16x32 bf16: lane l holds row l%16, element e -> K = e + 16*(l/16)
// B 32x16 bf16: lane l holds column l%16, element e -> K = e + 16*(l/16)
// ---------------------------------------------------------------------------
__device__ __forceinline__ v16bf load_a_lds(const __bf16* __restrict__ xr, int k0, int hl) {
  union { v16bf v; v8bf h[2]; } a;
  a.h[0] = *(const v8bf*)(xr + k0 + 16 * hl);
  a.h[1] = *(const v8bf*)(xr + k0 + 16 * hl + 8);
  return a.v;
}
__device__ __forceinline__ v16bf load_b_bf16(const __bf16* __restrict__ w, int k0) {
  union { v16bf v; v8bf h[2]; } b;
  b.h[0] = *(const v8bf*)(w + k0);
  b.h[1] = *(const v8bf*)(w + k0 + 8);
  return b.v;
}
__device__ __forceinline__ v16bf load_b_f32(const float* __restrict__ w, int k0) {
  v16bf b;
#pragma unroll
  for (int j = 0; j < 16; ++j) b[j] = f2bf(w[k0 + j]);
  return b;
}

// ---------------------------------------------------------------------------
// Expert / shared FFN: out[t] (+)= w[t] * ( silu(X Wg^T) * (X Wu^T) ) Wd^T
// 256 threads = 8 waves; each wave owns a column group and BOTH 16-row tiles,
// so every B fragment feeds two WMMAs (halves global weight traffic).
// SHARED=true : dense tokens, w=1, plain stores (initializes d_out).
// SHARED=false: gathered tokens of expert blockIdx.x, f32 global atomic adds.
// BF16W=true  : weights pre-converted to bf16 (two b128 loads per fragment).
// ---------------------------------------------------------------------------
template <bool SHARED, bool BF16W>
__global__ __launch_bounds__(256) void ffn_kernel(
    const float* __restrict__ x,
    const void* __restrict__ Wg, const void* __restrict__ Wu, const void* __restrict__ Wd,
    const float* __restrict__ combine, const int* __restrict__ counts,
    const int* __restrict__ lists, float* __restrict__ out)
{
  constexpr int XS = DD + 8;   // +16B row pad: spreads LDS banks, keeps 16B align
  constexpr int HS = II + 8;
  __shared__ __bf16 Xs[MBLK * XS];
  __shared__ __bf16 Hs[MBLK * HS];
  __shared__ int    toks[MBLK];
  __shared__ float  wts[MBLK];

  const int e   = SHARED ? 0 : blockIdx.x;
  const int mb  = blockIdx.y;
  const int cnt = SHARED ? TT : counts[e];
  if (mb * MBLK >= cnt) return;                 // uniform across block

  const int tid = threadIdx.x;
  if (tid < MBLK) {
    const int i = mb * MBLK + tid;
    int t; float w;
    if (SHARED)          { t = i;                 w = 1.f; }
    else if (i < cnt)    { t = lists[e * TT + i]; w = combine[(size_t)t * EE + e]; }
    else                 { t = 0;                 w = 0.f; }
    toks[tid] = t; wts[tid] = w;
  }
  __syncthreads();

  // Gather X tile -> bf16 LDS (float4 loads, 32 rows x 1024 cols)
  for (int idx = tid; idx < MBLK * (DD / 4); idx += 256) {
    const int r = idx >> 8;             // DD/4 == 256
    const int c = (idx & 255) * 4;
    const float4 v = *(const float4*)(x + (size_t)toks[r] * DD + c);
    __bf16* dst = &Xs[r * XS + c];
    dst[0] = f2bf(v.x); dst[1] = f2bf(v.y); dst[2] = f2bf(v.z); dst[3] = f2bf(v.w);
  }
  __syncthreads();

  const int wave = tid >> 5;            // 0..7: column group
  const int lane = tid & 31;
  const int nl   = lane & 15;           // A row-in-tile / B column / D column
  const int hl   = lane >> 4;           // lane half

  const size_t wbase = SHARED ? 0 : (size_t)e * II * DD;
  const size_t dbase = SHARED ? 0 : (size_t)e * DD * II;

  const __bf16* xr0 = &Xs[nl * XS];            // row tile 0: rows 0..15
  const __bf16* xr1 = &Xs[(16 + nl) * XS];     // row tile 1: rows 16..31

  // -------- Stage 1: H = silu(X Wg^T) * (X Wu^T), N = II (4 col-tiles/wave) --------
  for (int ct = wave * 4; ct < wave * 4 + 4; ++ct) {
    const int n0 = ct * 16;
    v8f accG0 = {}, accG1 = {}, accU0 = {}, accU1 = {};
    const size_t rowoff = wbase + (size_t)(n0 + nl) * DD + 16 * hl;
    const __bf16* gwb = (const __bf16*)Wg + rowoff;
    const __bf16* uwb = (const __bf16*)Wu + rowoff;
    const float*  gwf = (const float*)Wg + rowoff;
    const float*  uwf = (const float*)Wu + rowoff;
    for (int k0 = 0; k0 < DD; k0 += 32) {
      if (BF16W) __builtin_prefetch(gwb + k0 + 64, 0, 0);   // global_prefetch_b8
      else       __builtin_prefetch(gwf + k0 + 32, 0, 0);
      const v16bf a0 = load_a_lds(xr0, k0, hl);
      const v16bf a1 = load_a_lds(xr1, k0, hl);
      const v16bf bg = BF16W ? load_b_bf16(gwb, k0) : load_b_f32(gwf, k0);
      const v16bf bu = BF16W ? load_b_bf16(uwb, k0) : load_b_f32(uwf, k0);
      accG0 = __builtin_amdgcn_wmma_f32_16x16x32_bf16(false, a0, false, bg, (short)0, accG0, false, false);
      accG1 = __builtin_amdgcn_wmma_f32_16x16x32_bf16(false, a1, false, bg, (short)0, accG1, false, false);
      accU0 = __builtin_amdgcn_wmma_f32_16x16x32_bf16(false, a0, false, bu, (short)0, accU0, false, false);
      accU1 = __builtin_amdgcn_wmma_f32_16x16x32_bf16(false, a1, false, bu, (short)0, accU1, false, false);
    }
#pragma unroll
    for (int r = 0; r < 8; ++r) {                // M = r + 8*hl, N = nl
      const int m = r + 8 * hl;
      const float g0 = accG0[r], u0 = accU0[r];
      const float g1 = accG1[r], u1 = accU1[r];
      Hs[m * HS + n0 + nl]        = f2bf((g0 / (1.f + __expf(-g0))) * u0);
      Hs[(16 + m) * HS + n0 + nl] = f2bf((g1 / (1.f + __expf(-g1))) * u1);
    }
  }
  __syncthreads();

  // -------- Stage 2: Y = H Wd^T, N = DD (8 col-tiles/wave); scale + emit --------
  const __bf16* hr0 = &Hs[nl * HS];
  const __bf16* hr1 = &Hs[(16 + nl) * HS];
  for (int ct = wave * 8; ct < wave * 8 + 8; ++ct) {
    const int n0 = ct * 16;
    v8f acc0 = {}, acc1 = {};
    const size_t rowoff = dbase + (size_t)(n0 + nl) * II + 16 * hl;
    const __bf16* dwb = (const __bf16*)Wd + rowoff;
    const float*  dwf = (const float*)Wd + rowoff;
    for (int k0 = 0; k0 < II; k0 += 32) {
      const v16bf a0 = load_a_lds(hr0, k0, hl);
      const v16bf a1 = load_a_lds(hr1, k0, hl);
      const v16bf bd = BF16W ? load_b_bf16(dwb, k0) : load_b_f32(dwf, k0);
      acc0 = __builtin_amdgcn_wmma_f32_16x16x32_bf16(false, a0, false, bd, (short)0, acc0, false, false);
      acc1 = __builtin_amdgcn_wmma_f32_16x16x32_bf16(false, a1, false, bd, (short)0, acc1, false, false);
    }
#pragma unroll
    for (int r = 0; r < 8; ++r) {
      const int m = r + 8 * hl;
#pragma unroll
      for (int half = 0; half < 2; ++half) {
        const int   row = 16 * half + m;
        const float w   = wts[row];
        const float v   = (half ? acc1[r] : acc0[r]);
        float* dst = out + (size_t)toks[row] * DD + n0 + nl;
        if (SHARED)          *dst = v;                        // w == 1
        else if (w != 0.f)   unsafeAtomicAdd(dst, v * w);     // global_atomic_add_f32
      }
    }
  }
}

// ---------------------------------------------------------------------------
extern "C" void kernel_launch(void* const* d_in, const int* in_sizes, int n_in,
                              void* d_out, int out_size, void* d_ws, size_t ws_size,
                              hipStream_t stream) {
  const float* x  = (const float*)d_in[0];   // hidden_states [B,S,D]
  const float* gw = (const float*)d_in[1];   // gate_w        [E,D]
  const float* eg = (const float*)d_in[2];   // expert_gate   [E,I,D]
  const float* eu = (const float*)d_in[3];   // expert_up     [E,I,D]
  const float* ed = (const float*)d_in[4];   // expert_down   [E,D,I]
  const float* sg = (const float*)d_in[5];   // shared_gate_w [I,D]
  const float* su = (const float*)d_in[6];   // shared_up_w   [I,D]
  const float* sd = (const float*)d_in[7];   // shared_down_w [D,I]

  float* out    = (float*)d_out;                  // [T, D]
  float* logits = out + (size_t)TT * DD;          // [T, E]

  char*  ws      = (char*)d_ws;
  float* combine = (float*)ws;                                         // T*E f32
  int*   counts  = (int*)(ws + (size_t)TT * EE * sizeof(float));       // E i32
  int*   lists   = (int*)(ws + (size_t)TT * EE * sizeof(float) + 256); // E*T i32

  const size_t EID   = (size_t)EE * II * DD;   // elems per expert tensor
  const size_t SID   = (size_t)II * DD;        // elems per shared tensor
  size_t bfoff = (size_t)TT * EE * 4 + 256 + (size_t)EE * TT * 4;
  bfoff = (bfoff + 255) & ~(size_t)255;
  const size_t need = bfoff + (3 * EID + 3 * SID) * sizeof(__bf16);
  const bool use_bf16w = (ws_size >= need);

  hipMemsetAsync(counts, 0, EE * sizeof(int), stream);
  router_kernel<<<TT / 16, 32, 0, stream>>>(x, gw, logits, combine);
  build_lists_kernel<<<TT / 256, 256, 0, stream>>>(combine, counts, lists);

  if (use_bf16w) {
    __bf16* egb = (__bf16*)(ws + bfoff);
    __bf16* eub = egb + EID;
    __bf16* edb = eub + EID;
    __bf16* sgb = edb + EID;
    __bf16* sub = sgb + SID;
    __bf16* sdb = sub + SID;
    const int n4e = (int)(EID / 4), n4s = (int)(SID / 4);
    cvt_bf16_kernel<<<(n4e + 255) / 256, 256, 0, stream>>>(eg, egb, n4e);
    cvt_bf16_kernel<<<(n4e + 255) / 256, 256, 0, stream>>>(eu, eub, n4e);
    cvt_bf16_kernel<<<(n4e + 255) / 256, 256, 0, stream>>>(ed, edb, n4e);
    cvt_bf16_kernel<<<(n4s + 255) / 256, 256, 0, stream>>>(sg, sgb, n4s);
    cvt_bf16_kernel<<<(n4s + 255) / 256, 256, 0, stream>>>(su, sub, n4s);
    cvt_bf16_kernel<<<(n4s + 255) / 256, 256, 0, stream>>>(sd, sdb, n4s);
    ffn_kernel<true,  true><<<dim3(1,  TT / MBLK), 256, 0, stream>>>(x, sgb, sub, sdb, combine, counts, lists, out);
    ffn_kernel<false, true><<<dim3(EE, TT / MBLK), 256, 0, stream>>>(x, egb, eub, edb, combine, counts, lists, out);
  } else {
    ffn_kernel<true,  false><<<dim3(1,  TT / MBLK), 256, 0, stream>>>(x, sg, su, sd, combine, counts, lists, out);
    ffn_kernel<false, false><<<dim3(EE, TT / MBLK), 256, 0, stream>>>(x, eg, eu, ed, combine, counts, lists, out);
  }
}
